// PointNetSetAbstractionMsg_7516192768034
// MI455X (gfx1250) — matrix operand order repaired and checked by reading, not compiled, not run on hardware
//
#include <hip/hip_runtime.h>

// ---------------------------------------------------------------------------
// PointNet++ Set-Abstraction (MSG) for MI455X / gfx1250 (wave32, WMMA).
//
// Roofline-driven structure: batch-stat BN forces a global reduction between
// layers, but storing pre-BN f32 activations is the bandwidth bottleneck.
// WMMA FLOPs (~40 GFLOP) are nearly free vs 23.3 TB/s HBM, so we run each
// layer TWICE instead of spilling f32:
//   pass 1: WMMA GEMM -> per-channel sum/ssq only (LDS ds_add_f32 +
//           global atomic_add_f32)
//   pass 2: recompute WMMA GEMM -> fused BN(scale,shift)+ReLU -> f16 store
//           (last layer: fused max-over-K via LDS atomicMax -> d_out)
// Second-pass X re-reads mostly hit the 192 MB global L2.
//
// WMMA tile: each wave computes 32(M) x 32(O) via 4x v_wmma_f32_16x16x32_f16
// per 32-wide K step. Fragment layouts used here (contiguous K, same order for A and B):
//   A (16x32 f16): lane<16: row l, K[0..16); lane>=16: row l-16,
//                  K[16..32)            -> two contiguous 16B loads
//   B (32x16 f16): lane<16: col l, K[0..16); lane>=16: col l-16, K[16..32)
//                                        -> one contiguous 32B load
//   C/D: VGPR r -> M = r + (lane>=16 ? 8 : 0), N = lane%16
// Conv bias cancels exactly under batch-stat BN and is skipped.
// ---------------------------------------------------------------------------

typedef __attribute__((ext_vector_type(16))) _Float16 v16h;
typedef __attribute__((ext_vector_type(8)))  _Float16 v8h;
typedef __attribute__((ext_vector_type(8)))  float    v8f;

#define B_ 16
#define N_ 4096
#define S_ 512
#define BN_EPS 1e-5f

// ---------------- Farthest point sampling (1 block per batch) --------------
__global__ __launch_bounds__(256) void fps_kernel(
    const float* __restrict__ xyz,   // (B,3,N)
    int*   __restrict__ fpsIdx,      // (B,S)
    float* __restrict__ newXyz,      // (B,S,3)  scratch, for ball query
    float* __restrict__ outXyz)      // (B,3,S)  first output tensor
{
  __shared__ float dist[N_];
  __shared__ float redV[256];
  __shared__ int   redI[256];
  __shared__ float cen[3];
  __shared__ int   curIdx;
  const int b = blockIdx.x, t = threadIdx.x;
  const float* px = xyz + (size_t)b * 3 * N_;
  for (int i = t; i < N_; i += 256) dist[i] = 1e10f;
  if (t == 0) curIdx = 0;
  __syncthreads();
  for (int it = 0; it < S_; ++it) {
    if (t == 0) {
      int f = curIdx;
      fpsIdx[b * S_ + it] = f;
      float x = px[f], y = px[N_ + f], z = px[2 * N_ + f];
      cen[0] = x; cen[1] = y; cen[2] = z;
      size_t gs = (size_t)b * S_ + it;
      newXyz[gs * 3 + 0] = x; newXyz[gs * 3 + 1] = y; newXyz[gs * 3 + 2] = z;
      outXyz[(size_t)b * 3 * S_ + 0 * S_ + it] = x;
      outXyz[(size_t)b * 3 * S_ + 1 * S_ + it] = y;
      outXyz[(size_t)b * 3 * S_ + 2 * S_ + it] = z;
    }
    __syncthreads();
    const float cx = cen[0], cy = cen[1], cz = cen[2];
    float best = -1.0f; int bi = 0;
    for (int i = t; i < N_; i += 256) {
      float dx = px[i] - cx, dy = px[N_ + i] - cy, dz = px[2 * N_ + i] - cz;
      float d  = dx * dx + dy * dy + dz * dz;
      float dm = fminf(dist[i], d);
      dist[i] = dm;
      if (dm > best) { best = dm; bi = i; }
    }
    redV[t] = best; redI[t] = bi;
    __syncthreads();
    for (int s = 128; s > 0; s >>= 1) {
      if (t < s && redV[t + s] > redV[t]) { redV[t] = redV[t + s]; redI[t] = redI[t + s]; }
      __syncthreads();
    }
    if (t == 0) curIdx = redI[0];
    __syncthreads();
  }
}

// ------------- Ball query + group + f16 pack (thread per (b,s)) ------------
__global__ __launch_bounds__(256) void group_kernel(
    const float* __restrict__ xyz,     // (B,3,N)
    const float* __restrict__ pts,     // (B,3,N)
    const float* __restrict__ newXyz,  // (B,S,3)
    _Float16* __restrict__ X,          // (B*S*K, 32) f16, zero-padded channels
    int K, float r2)
{
  int gs = blockIdx.x * blockDim.x + threadIdx.x;   // b*S + s
  if (gs >= B_ * S_) return;
  const int b = gs >> 9;
  const float* px = xyz + (size_t)b * 3 * N_;
  const float* pp = pts + (size_t)b * 3 * N_;
  const float cx = newXyz[(size_t)gs * 3 + 0];
  const float cy = newXyz[(size_t)gs * 3 + 1];
  const float cz = newXyz[(size_t)gs * 3 + 2];
  const size_t rowBase = (size_t)gs * K;
  int cnt = 0;
  float f0 = 0, f1 = 0, f2 = 0, f3 = 0, f4 = 0, f5 = 0;
  for (int n = 0; n < N_ && cnt < K; ++n) {
    float dx = px[n] - cx, dy = px[N_ + n] - cy, dz = px[2 * N_ + n] - cz;
    if (dx * dx + dy * dy + dz * dz <= r2) {
      float p0 = pp[n], p1 = pp[N_ + n], p2 = pp[2 * N_ + n];
      if (cnt == 0) { f0 = p0; f1 = p1; f2 = p2; f3 = dx; f4 = dy; f5 = dz; }
      _Float16* row = X + (rowBase + cnt) * 32;
      row[0] = (_Float16)p0; row[1] = (_Float16)p1; row[2] = (_Float16)p2;
      row[3] = (_Float16)dx; row[4] = (_Float16)dy; row[5] = (_Float16)dz;
      #pragma unroll
      for (int c = 6; c < 32; ++c) row[c] = (_Float16)0.0f;
      ++cnt;
    }
  }
  if (cnt == 0) {  // degenerate: JAX clamps index N -> N-1
    int n = N_ - 1;
    f0 = pp[n]; f1 = pp[N_ + n]; f2 = pp[2 * N_ + n];
    f3 = px[n] - cx; f4 = px[N_ + n] - cy; f5 = px[2 * N_ + n] - cz;
  }
  for (int k = cnt; k < K; ++k) {   // pad with first in-radius point
    _Float16* row = X + (rowBase + k) * 32;
    row[0] = (_Float16)f0; row[1] = (_Float16)f1; row[2] = (_Float16)f2;
    row[3] = (_Float16)f3; row[4] = (_Float16)f4; row[5] = (_Float16)f5;
    #pragma unroll
    for (int c = 6; c < 32; ++c) row[c] = (_Float16)0.0f;
  }
}

// ------------------- weights f32 (O,C) -> f16 (O,Cpad) ---------------------
__global__ void wconvert_kernel(const float* __restrict__ W,
                                _Float16* __restrict__ Wh, int O, int C, int Cpad)
{
  int i = blockIdx.x * blockDim.x + threadIdx.x;
  if (i < O * Cpad) {
    int o = i / Cpad, c = i % Cpad;
    Wh[i] = (_Float16)((c < C) ? W[o * C + c] : 0.0f);
  }
}

__global__ void zero_kernel(float* __restrict__ p, int n)
{
  int i = blockIdx.x * blockDim.x + threadIdx.x;
  if (i < n) p[i] = 0.0f;
}

// ----------------- shared WMMA 32x32 tile mainloop -------------------------
__device__ __forceinline__ void wmma_tile_32x32(
    const _Float16* __restrict__ X, const _Float16* __restrict__ Wh,
    size_t mBase, int oBase, int Cpad, int lhalf, int lhi,
    v8f& acc00, v8f& acc01, v8f& acc10, v8f& acc11)
{
  const size_t rowA0 = mBase + lhalf;
  const size_t rowA1 = rowA0 + 16;
  const int aK = lhi * 16;    // A frag K phase: contiguous K[16h..16h+16), matches B
  const int bK = lhi * 16;    // B frag K phase: contiguous K[16h..16h+16), matches A
  const size_t wRow0 = (size_t)(oBase + lhalf) * Cpad;
  const size_t wRow1 = wRow0 + (size_t)16 * Cpad;

  for (int c = 0; c < Cpad; c += 32) {
    const _Float16* a0p = X + rowA0 * Cpad + c + aK;
    const _Float16* a1p = X + rowA1 * Cpad + c + aK;
    v8h a0lo = *(const v8h*)(a0p);
    v8h a0hi = *(const v8h*)(a0p + 8);
    v8h a1lo = *(const v8h*)(a1p);
    v8h a1hi = *(const v8h*)(a1p + 8);
    v16h a0 = __builtin_shufflevector(a0lo, a0hi, 0,1,2,3,4,5,6,7,8,9,10,11,12,13,14,15);
    v16h a1 = __builtin_shufflevector(a1lo, a1hi, 0,1,2,3,4,5,6,7,8,9,10,11,12,13,14,15);
    v16h b0 = *(const v16h*)(Wh + wRow0 + c + bK);
    v16h b1 = *(const v16h*)(Wh + wRow1 + c + bK);
    acc00 = __builtin_amdgcn_wmma_f32_16x16x32_f16(false, a0, false, b0, (short)0, acc00, false, false);
    acc01 = __builtin_amdgcn_wmma_f32_16x16x32_f16(false, a0, false, b1, (short)0, acc01, false, false);
    acc10 = __builtin_amdgcn_wmma_f32_16x16x32_f16(false, a1, false, b0, (short)0, acc10, false, false);
    acc11 = __builtin_amdgcn_wmma_f32_16x16x32_f16(false, a1, false, b1, (short)0, acc11, false, false);
  }
}

// ---------- Pass 1: WMMA GEMM -> per-channel sum / sum-of-squares ----------
// grid = (O/32, M/256), block = 256 (8 waves).
__global__ __launch_bounds__(256) void gemm_stats_kernel(
    const _Float16* __restrict__ X,   // (M, Cpad) f16
    const _Float16* __restrict__ Wh,  // (O, Cpad) f16
    float* __restrict__ gsum, float* __restrict__ gssq,
    int Cpad, int O)
{
  __shared__ float sSum[32], sSsq[32];
  const int tid = threadIdx.x;
  if (tid < 32) { sSum[tid] = 0.0f; sSsq[tid] = 0.0f; }
  __syncthreads();

  const int wave  = tid >> 5;
  const int lane  = tid & 31;
  const int lhalf = lane & 15;
  const int lhi   = lane >> 4;
  const int    oBase = blockIdx.x * 32;
  const size_t mBase = ((size_t)blockIdx.y * 8 + wave) * 32;

  v8f acc00 = {}, acc01 = {}, acc10 = {}, acc11 = {};
  wmma_tile_32x32(X, Wh, mBase, oBase, Cpad, lhalf, lhi, acc00, acc01, acc10, acc11);

  float s1a = 0, s2a = 0, s1b = 0, s2b = 0;
  #pragma unroll
  for (int r = 0; r < 8; ++r) {
    float v00 = acc00[r], v01 = acc01[r], v10 = acc10[r], v11 = acc11[r];
    s1a += v00 + v10;  s2a += v00 * v00 + v10 * v10;
    s1b += v01 + v11;  s2b += v01 * v01 + v11 * v11;
  }
  atomicAdd(&sSum[lhalf], s1a);       atomicAdd(&sSsq[lhalf], s2a);
  atomicAdd(&sSum[16 + lhalf], s1b);  atomicAdd(&sSsq[16 + lhalf], s2b);
  __syncthreads();
  if (tid < 32) {
    atomicAdd(&gsum[oBase + tid], sSum[tid]);
    atomicAdd(&gssq[oBase + tid], sSsq[tid]);
  }
}

// ----------------- fold batch stats + gamma/beta into scale/shift ----------
__global__ void bn_finalize_kernel(const float* __restrict__ gsum,
                                   const float* __restrict__ gssq,
                                   const float* __restrict__ gamma,
                                   const float* __restrict__ beta,
                                   float* __restrict__ scale,
                                   float* __restrict__ shift,
                                   int O, float invM)
{
  int o = blockIdx.x * blockDim.x + threadIdx.x;
  if (o < O) {
    float m  = gsum[o] * invM;
    float v  = gssq[o] * invM - m * m;
    float sc = gamma[o] * rsqrtf(v + BN_EPS);
    scale[o] = sc;
    shift[o] = beta[o] - m * sc;
  }
}

// ------- Pass 2 (inner layers): recompute GEMM, fused BN+ReLU -> f16 -------
__global__ __launch_bounds__(256) void gemm_apply_kernel(
    const _Float16* __restrict__ X,
    const _Float16* __restrict__ Wh,
    const float* __restrict__ scale, const float* __restrict__ shift,
    _Float16* __restrict__ Xout,      // (M, O) f16 next-layer input
    int Cpad, int O)
{
  const int tid = threadIdx.x;
  const int wave  = tid >> 5;
  const int lane  = tid & 31;
  const int lhalf = lane & 15;
  const int lhi   = lane >> 4;
  const int    oBase = blockIdx.x * 32;
  const size_t mBase = ((size_t)blockIdx.y * 8 + wave) * 32;

  v8f acc00 = {}, acc01 = {}, acc10 = {}, acc11 = {};
  wmma_tile_32x32(X, Wh, mBase, oBase, Cpad, lhalf, lhi, acc00, acc01, acc10, acc11);

  const size_t rA = mBase + (lhi ? 8 : 0);
  const size_t rB = rA + 16;
  const int col0 = oBase + lhalf;
  const int col1 = col0 + 16;
  const float sc0 = scale[col0], sh0 = shift[col0];
  const float sc1 = scale[col1], sh1 = shift[col1];
  #pragma unroll
  for (int r = 0; r < 8; ++r) {
    Xout[(rA + r) * O + col0] = (_Float16)fmaxf(acc00[r] * sc0 + sh0, 0.0f);
    Xout[(rA + r) * O + col1] = (_Float16)fmaxf(acc01[r] * sc1 + sh1, 0.0f);
    Xout[(rB + r) * O + col0] = (_Float16)fmaxf(acc10[r] * sc0 + sh0, 0.0f);
    Xout[(rB + r) * O + col1] = (_Float16)fmaxf(acc11[r] * sc1 + sh1, 0.0f);
  }
}

// -- Pass 2 (last layer): GEMM + BN + ReLU + max-over-K fused, -> d_out -----
// K in {16,32,128} divides the 256 rows a block owns, so every (b,s) group is
// block-local: reduce with LDS atomicMax on int bits (valid: values >= 0).
__global__ __launch_bounds__(256) void gemm_apply_max_kernel(
    const _Float16* __restrict__ X,
    const _Float16* __restrict__ Wh,
    const float* __restrict__ scale, const float* __restrict__ shift,
    float* __restrict__ outPts,       // (B, 320, S)
    int Cpad, int O, int K, int chOff)
{
  __shared__ int smax[16][32];        // [group-in-block][col-in-block]
  const int tid = threadIdx.x;
  const int gpb = 256 / K;            // groups per block (<= 16)
  for (int i = tid; i < gpb * 32; i += 256) smax[i >> 5][i & 31] = 0;
  __syncthreads();

  const int wave  = tid >> 5;
  const int lane  = tid & 31;
  const int lhalf = lane & 15;
  const int lhi   = lane >> 4;
  const int    oBase = blockIdx.x * 32;
  const size_t rowBlk = (size_t)blockIdx.y * 256;
  const size_t mBase  = rowBlk + (size_t)wave * 32;

  v8f acc00 = {}, acc01 = {}, acc10 = {}, acc11 = {};
  wmma_tile_32x32(X, Wh, mBase, oBase, Cpad, lhalf, lhi, acc00, acc01, acc10, acc11);

  const int rowA = wave * 32 + (lhi ? 8 : 0);   // row within block
  const int rowB = rowA + 16;
  const int col0 = oBase + lhalf;
  const int col1 = col0 + 16;
  const float sc0 = scale[col0], sh0 = shift[col0];
  const float sc1 = scale[col1], sh1 = shift[col1];
  #pragma unroll
  for (int r = 0; r < 8; ++r) {
    int gA = (rowA + r) / K, gB = (rowB + r) / K;
    float v00 = fmaxf(acc00[r] * sc0 + sh0, 0.0f);
    float v01 = fmaxf(acc01[r] * sc1 + sh1, 0.0f);
    float v10 = fmaxf(acc10[r] * sc0 + sh0, 0.0f);
    float v11 = fmaxf(acc11[r] * sc1 + sh1, 0.0f);
    atomicMax(&smax[gA][lhalf],      __float_as_int(v00));
    atomicMax(&smax[gA][16 + lhalf], __float_as_int(v01));
    atomicMax(&smax[gB][lhalf],      __float_as_int(v10));
    atomicMax(&smax[gB][16 + lhalf], __float_as_int(v11));
  }
  __syncthreads();
  for (int i = tid; i < gpb * 32; i += 256) {
    int g = i >> 5, cc = i & 31;
    int gs = (int)(rowBlk / K) + g;             // b*S + s
    int b = gs >> 9, s = gs & (S_ - 1);
    outPts[((long)b * 320 + chOff + oBase + cc) * S_ + s] = __int_as_float(smax[g][cc]);
  }
}

// ---------------------------------------------------------------------------
extern "C" void kernel_launch(void* const* d_in, const int* in_sizes, int n_in,
                              void* d_out, int out_size, void* d_ws, size_t ws_size,
                              hipStream_t stream)
{
  (void)in_sizes; (void)n_in; (void)out_size; (void)ws_size;
  const float* xyz = (const float*)d_in[0];
  const float* pts = (const float*)d_in[1];
  // params flattened: d_in[2 + (br*3+ly)*4 + {W,b,gamma,beta}]

  static const int   KS[3]       = {16, 32, 128};
  static const float R2[3]       = {0.01f, 0.04f, 0.16f};
  static const int   OUTCH[3][3] = {{32, 32, 64}, {64, 64, 128}, {64, 96, 128}};
  static const int   CHOFF[3]    = {0, 64, 192};

  char* ws = (char*)d_ws;
  size_t off = 0;
  auto alloc = [&](size_t bytes) -> char* {
    char* p = ws + off;
    off += (bytes + 255) & ~(size_t)255;
    return p;
  };

  int*      fpsIdx = (int*)   alloc((size_t)B_ * S_ * sizeof(int));
  float*    newXyz = (float*) alloc((size_t)B_ * S_ * 3 * sizeof(float));
  float*    gsum   = (float*) alloc(128 * sizeof(float));
  float*    gssq   = (float*) alloc(128 * sizeof(float));
  float*    scale  = (float*) alloc(128 * sizeof(float));
  float*    shift  = (float*) alloc(128 * sizeof(float));
  _Float16* Wh[3][3];
  for (int br = 0; br < 3; ++br)
    for (int ly = 0; ly < 3; ++ly)
      Wh[br][ly] = (_Float16*)alloc((size_t)128 * 128 * sizeof(_Float16));
  const size_t MAXM = (size_t)B_ * S_ * 128;                 // 1,048,576 rows
  _Float16* Xa = (_Float16*)alloc(MAXM * 128 * sizeof(_Float16)); // 256 MB
  _Float16* Xb = (_Float16*)alloc(MAXM * 128 * sizeof(_Float16)); // 256 MB

  float* outXyz = (float*)d_out;
  float* outPts = (float*)d_out + (size_t)B_ * 3 * S_;

  // 1) FPS (also writes new_xyz output)
  fps_kernel<<<B_, 256, 0, stream>>>(xyz, fpsIdx, newXyz, outXyz);

  // 2) Convert all 9 weight matrices to padded f16 once
  for (int br = 0; br < 3; ++br) {
    int cin = 6, cpad = 32;
    for (int ly = 0; ly < 3; ++ly) {
      const float* W = (const float*)d_in[2 + (br * 3 + ly) * 4 + 0];
      int O = OUTCH[br][ly];
      int n = O * cpad;
      wconvert_kernel<<<(n + 255) / 256, 256, 0, stream>>>(W, Wh[br][ly], O, cin, cpad);
      cin = O; cpad = O;
    }
  }

  // 3) Per-radius branch: group -> (stats GEMM -> finalize -> apply GEMM) x3
  for (int br = 0; br < 3; ++br) {
    const int    K = KS[br];
    const size_t M = (size_t)B_ * S_ * K;
    group_kernel<<<(B_ * S_ + 255) / 256, 256, 0, stream>>>(xyz, pts, newXyz, Xa, K, R2[br]);
    _Float16* Xin  = Xa;
    _Float16* Xout = Xb;
    int cpad = 32;
    for (int ly = 0; ly < 3; ++ly) {
      const int O = OUTCH[br][ly];
      const float* gamma = (const float*)d_in[2 + (br * 3 + ly) * 4 + 2];
      const float* beta  = (const float*)d_in[2 + (br * 3 + ly) * 4 + 3];
      zero_kernel<<<1, 128, 0, stream>>>(gsum, 128);
      zero_kernel<<<1, 128, 0, stream>>>(gssq, 128);
      dim3 grid(O / 32, (unsigned)(M / 256));
      gemm_stats_kernel<<<grid, 256, 0, stream>>>(Xin, Wh[br][ly], gsum, gssq, cpad, O);
      bn_finalize_kernel<<<1, 128, 0, stream>>>(gsum, gssq, gamma, beta, scale, shift,
                                                O, 1.0f / (float)M);
      if (ly < 2) {
        gemm_apply_kernel<<<grid, 256, 0, stream>>>(Xin, Wh[br][ly], scale, shift,
                                                    Xout, cpad, O);
        _Float16* t = Xin; Xin = Xout; Xout = t;
      } else {
        gemm_apply_max_kernel<<<grid, 256, 0, stream>>>(Xin, Wh[br][ly], scale, shift,
                                                        outPts, cpad, O, K, CHOFF[br]);
      }
      cpad = O;
    }
  }
}
